// GraphAttentionLayer_8564164789092
// MI455X (gfx1250) — matrix operand order repaired and checked by reading, not compiled, not run on hardware
//
#include <hip/hip_runtime.h>
#include <hip/hip_bf16.h>

// GAT layer, MI455X (gfx1250), wave32, bf16 WMMA (v_wmma_f32_16x16x32_bf16).
// B=8, N=2048, F_IN=256, F_OUT=128.

#define B_    8
#define N_    2048
#define FIN   256
#define FOUT  128
#define ALPHA 0.2f
#define NEG_BIG (-9.0e15f)

typedef __attribute__((ext_vector_type(8)))  float  v8f;
typedef __attribute__((ext_vector_type(16))) __bf16 v16bf;

__device__ __forceinline__ float leaky(float x) { return x > 0.f ? x : ALPHA * x; }

// ---------------------------------------------------------------------------
// Kernel 0: swizzle W (fp32 [256][128]) -> bf16 in WMMA B-fragment layout:
//   WB[((kk*8 + t)*32 + lane)*16 + j]  where for lane L: n = L&15,
//   k = kk*32 + (L&16) + j   (B matrix is K x N = 32 x 16 per WMMA tile t)
// ---------------------------------------------------------------------------
__global__ __launch_bounds__(256) void gat_swizzleW(const float* __restrict__ W,
                                                    __bf16* __restrict__ WB) {
  const int idx = blockIdx.x * 256 + threadIdx.x;   // 0 .. 32767
  const int k = idx >> 7;                           // 0 .. 255
  const int c = idx & 127;                          // 0 .. 127
  const int kk = k >> 5, jpos = k & 31;
  const int t = c >> 4, n = c & 15;
  const int Lp = n + (jpos & 16);
  const int jp = jpos & 15;
  WB[((kk * 8 + t) * 32 + Lp) * 16 + jp] = (__bf16)W[idx];
}

// ---------------------------------------------------------------------------
// Kernel 1: Wh = h @ W  via bf16 WMMA.  Each wave: 16 rows x 128 cols.
// Also computes s1 = Wh@a1, s2 = Wh@a2, and stores Wh as bf16 pre-swizzled
// into the B-fragment layout indexed by j-tile (for the attention GEMM):
//   WhB[b][jt][t][lane][j],  jt = row/32, element (lane,j) <-> row jt*32+(lane&16)+j,
//   col t*16 + (lane&15).
// ---------------------------------------------------------------------------
__global__ __launch_bounds__(256) void gat_project(
    const float* __restrict__ h, const __bf16* __restrict__ WB,
    const float* __restrict__ a,
    __bf16* __restrict__ WhB, float* __restrict__ s1, float* __restrict__ s2) {
  const int lane  = threadIdx.x & 31;
  const int wave  = threadIdx.x >> 5;
  const int blk   = blockIdx.x;
  const int b     = blk >> 4;                 // 16 blocks per batch
  const int i0    = (blk & 15) * 128 + wave * 16;
  const int n     = lane & 15;
  const int base8 = (lane >> 4) << 3;         // 0 or 8

  const float* hrow = h + ((size_t)(b * N_ + i0 + n)) * FIN;  // A row m = lane&15

  v8f z = {};
  v8f acc[8];
#pragma unroll
  for (int t = 0; t < 8; ++t) acc[t] = z;

#pragma unroll
  for (int kk = 0; kk < 8; ++kk) {            // K = 256 in steps of 32
    v16bf af;
#pragma unroll
    for (int g = 0; g < 8; ++g) {
      // A-frag k order used here (contiguous, same as WB): vgpr g holds k = 2*base8 + 2*g + {0,1}
      const int k = kk * 32 + (base8 << 1) + (g << 1);
      const float2 hv = *(const float2*)(hrow + k);
      af[2 * g]     = (__bf16)hv.x;
      af[2 * g + 1] = (__bf16)hv.y;
    }
#pragma unroll
    for (int t = 0; t < 8; ++t) {
      const v16bf bf = *(const v16bf*)(WB + ((kk * 8 + t) * 32 + lane) * 16);
      acc[t] = __builtin_amdgcn_wmma_f32_16x16x32_bf16(
          false, af, false, bf, (short)0, acc[t], false, false);
    }
  }

  // s1/s2: C layout: lane holds col n for rows m = base8 + r.
  float p1[8], p2[8];
#pragma unroll
  for (int r = 0; r < 8; ++r) { p1[r] = 0.f; p2[r] = 0.f; }
#pragma unroll
  for (int t = 0; t < 8; ++t) {
    const float a1v = a[t * 16 + n];
    const float a2v = a[FOUT + t * 16 + n];
#pragma unroll
    for (int r = 0; r < 8; ++r) {
      p1[r] += acc[t][r] * a1v;
      p2[r] += acc[t][r] * a2v;
    }
  }
#pragma unroll
  for (int off = 1; off < 16; off <<= 1) {
#pragma unroll
    for (int r = 0; r < 8; ++r) {
      p1[r] += __shfl_xor(p1[r], off, 32);
      p2[r] += __shfl_xor(p2[r], off, 32);
    }
  }
  if (n == 0) {
#pragma unroll
    for (int r = 0; r < 8; ++r) {
      s1[b * N_ + i0 + base8 + r] = p1[r];
      s2[b * N_ + i0 + base8 + r] = p2[r];
    }
  }

  // Store Wh bf16 into B-fragment layout for the attention GEMM.
  const size_t woff = (size_t)b * (N_ / 32) * 4096;
#pragma unroll
  for (int t = 0; t < 8; ++t) {
#pragma unroll
    for (int r = 0; r < 8; ++r) {
      const int rg = i0 + base8 + r;          // global row of Wh (= K index later)
      const int jt = rg >> 5, jl = rg & 31;
      const int Lp = n + (jl & 16);
      const int jp = jl & 15;
      WhB[woff + ((size_t)jt * 8 + t) * 512 + Lp * 16 + jp] = (__bf16)acc[t][r];
    }
  }
}

// ---------------------------------------------------------------------------
// Kernel 2: flash-style fused mask + softmax + (attention @ Wh).
// Each wave owns 16 query rows; loops over j in chunks of 32 (WMMA K).
// Scores are built directly in A-frag coordinates; online softmax stats per
// row live on lanes m and m+16 (combined with shfl_xor 16).
// ---------------------------------------------------------------------------
__global__ __launch_bounds__(128) void gat_attend(
    const int* __restrict__ adj, const __bf16* __restrict__ WhB,
    const float* __restrict__ s1g, const float* __restrict__ s2g,
    float* __restrict__ out) {
  const int lane  = threadIdx.x & 31;
  const int wave  = threadIdx.x >> 5;
  const int blk   = blockIdx.x;
  const int b     = blk >> 5;                 // 32 blocks per batch
  const int i0    = (blk & 31) * 64 + wave * 16;
  const int n     = lane & 15;
  const int base8 = (lane >> 4) << 3;

  const float  s1v    = s1g[b * N_ + i0 + n];  // this lane's query row = n
  const float* s2b    = s2g + b * N_;
  const int*   adjrow = adj + ((size_t)(b * N_ + i0 + n)) * N_;
  const __bf16* whb   = WhB + (size_t)b * (N_ / 32) * 4096;

  v8f z = {};
  v8f acc[8];
#pragma unroll
  for (int t = 0; t < 8; ++t) acc[t] = z;
  float mrun = -__builtin_inff();
  float lrun = 0.f;

  for (int jt = 0; jt < N_ / 32; ++jt) {
    const int j0 = jt * 32;
    if (jt + 1 < N_ / 32) __builtin_prefetch(adjrow + j0 + 32, 0, 0);  // global_prefetch_b8

    // Build e in A-fragment coordinates: elem pair (2g,2g+1) <-> k offsets.
    float e[16];
    float lmax = NEG_BIG;
#pragma unroll
    for (int g = 0; g < 8; ++g) {
      const int k = (base8 << 1) + (g << 1);
      const int2   ad = *(const int2*)(adjrow + j0 + k);
      const float2 sv = *(const float2*)(s2b + j0 + k);
      float x0 = leaky(s1v + sv.x); if (ad.x <= 0) x0 = NEG_BIG;
      float x1 = leaky(s1v + sv.y); if (ad.y <= 0) x1 = NEG_BIG;
      e[2 * g] = x0; e[2 * g + 1] = x1;
      lmax = fmaxf(lmax, fmaxf(x0, x1));
    }
    lmax = fmaxf(lmax, __shfl_xor(lmax, 16, 32));   // full row (32 j's) max
    const float mnew  = fmaxf(mrun, lmax);
    const float scale = __expf(mrun - mnew);        // first iter: exp(-inf)=0

    float rsum = 0.f;
    v16bf pf;
#pragma unroll
    for (int q = 0; q < 16; ++q) {
      const float p = __expf(e[q] - mnew);
      rsum += p;
      pf[q] = (__bf16)p;
    }
    rsum += __shfl_xor(rsum, 16, 32);
    lrun = lrun * scale + rsum;
    mrun = mnew;

    // Rescale accumulators: row m = base8 + r, its scale lives on lane (base8+r).
#pragma unroll
    for (int r = 0; r < 8; ++r) {
      const float sc = __shfl(scale, base8 + r, 32);
#pragma unroll
      for (int t = 0; t < 8; ++t) acc[t][r] *= sc;
    }

    // acc[t] += P(16x32) @ Wh[j-chunk](32x16) for each output tile t.
    const __bf16* bp = whb + (size_t)jt * 4096;
#pragma unroll
    for (int t = 0; t < 8; ++t) {
      const v16bf bf = *(const v16bf*)(bp + (t * 32 + lane) * 16);
      acc[t] = __builtin_amdgcn_wmma_f32_16x16x32_bf16(
          false, pf, false, bf, (short)0, acc[t], false, false);
    }
  }

  // Normalize by 1/l and store fp32 output.
  const float inv = 1.f / lrun;
  float* orow = out + ((size_t)(b * N_ + i0)) * FOUT;
#pragma unroll
  for (int r = 0; r < 8; ++r) {
    const float sc = __shfl(inv, base8 + r, 32);
#pragma unroll
    for (int t = 0; t < 8; ++t) {
      orow[(base8 + r) * FOUT + t * 16 + n] = acc[t][r] * sc;
    }
  }
}

// ---------------------------------------------------------------------------
extern "C" void kernel_launch(void* const* d_in, const int* in_sizes, int n_in,
                              void* d_out, int out_size, void* d_ws, size_t ws_size,
                              hipStream_t stream) {
  const float* h   = (const float*)d_in[0];
  const int*   adj = (const int*)d_in[1];
  const float* W   = (const float*)d_in[2];
  const float* a   = (const float*)d_in[3];
  float* out = (float*)d_out;

  char* ws = (char*)d_ws;
  __bf16* WhB = (__bf16*)ws;                                 // 8*2048*128*2 = 4 MiB
  float*  s1  = (float*)(ws + (size_t)B_ * N_ * FOUT * 2);   // 64 KiB
  float*  s2  = s1 + B_ * N_;                                // 64 KiB
  __bf16* WB  = (__bf16*)(s2 + B_ * N_);                     // 64 KiB swizzled W

  gat_swizzleW<<<(FIN * FOUT) / 256, 256, 0, stream>>>(W, WB);
  gat_project<<<(B_ * N_) / 128, 256, 0, stream>>>(h, WB, a, WhB, s1, s2);
  gat_attend<<<(B_ * N_) / 64, 128, 0, stream>>>(adj, WhB, s1, s2, out);
}
